// CharRNN_2396591751554
// MI455X (gfx1250) — matrix operand order repaired and checked by reading, not compiled, not run on hardware
//
#include <hip/hip_runtime.h>
#include <hip/hip_bf16.h>

// ---------------- problem constants ----------------
#define Vv 256
#define Ee 512
#define Hh 1024
#define Bb 64
#define Ss 512
#define G4H 4096        // 4*H
#define BS 32768        // B*S

typedef __attribute__((ext_vector_type(16))) __bf16 bf16x16;
typedef __attribute__((ext_vector_type(8)))  __bf16 bf16x8;
typedef __attribute__((ext_vector_type(8)))  float  f32x8;

static __device__ __forceinline__ f32x8 wmma_bf16(bf16x16 a, bf16x16 b, f32x8 c) {
  // D = A(16x32 bf16) * B(32x16 bf16) + C(16x16 f32)
  return __builtin_amdgcn_wmma_f32_16x16x32_bf16(false, a, false, b, (short)0, c, false, false);
}

// A-fragment (16x32, row-major [M,K] memory): lane m=lane%16, half=lane/16
//   elems 0..7  = A[m][k0 + half*8 .. +7]
//   elems 8..15 = A[m][k0 + 16 + half*8 .. +7]
static __device__ __forceinline__ bf16x16 load_a_frag(const __bf16* row, int k0, int half) {
  bf16x8 lo = *(const bf16x8*)(row + k0 + half * 8);
  bf16x8 hi = *(const bf16x8*)(row + k0 + 16 + half * 8);
  return __builtin_shufflevector(lo, hi, 0,1,2,3,4,5,6,7,8,9,10,11,12,13,14,15);
}

// B-fragment (32x16). We compute act @ W^T, so B[k][n] = W[n][k]; W stored [N,K] row-major.
//   lane n=lane%16, half=lane/16: elems 0..7 = W[n][k0 + half*8 .. +7], elems 8..15 = W[n][k0 + 16 + half*8 .. +7]
static __device__ __forceinline__ bf16x16 load_b_frag(const __bf16* row, int k0, int half) {
  bf16x8 lo = *(const bf16x8*)(row + k0 + half * 8);
  bf16x8 hi = *(const bf16x8*)(row + k0 + 16 + half * 8);
  return __builtin_shufflevector(lo, hi, 0,1,2,3,4,5,6,7,8,9,10,11,12,13,14,15);
}

static __device__ __forceinline__ float sigmoidf_(float x) {
  return 1.0f / (1.0f + __expf(-x));
}

// ---------------- utility kernels ----------------
__global__ void zero_counters_kernel(unsigned* cnt) {
  if (threadIdx.x < 2) cnt[threadIdx.x] = 0u;
}

__global__ void f32_to_bf16_kernel(const float* __restrict__ src, __bf16* __restrict__ dst, int n) {
  for (int i = blockIdx.x * blockDim.x + threadIdx.x; i < n; i += gridDim.x * blockDim.x)
    dst[i] = (__bf16)src[i];
}

// ---------------- generic bf16 WMMA GEMM:  C[M,N] = A@W^T + b1 (+ b2) ----------------
// A: bf16 [M,K] row-major (optionally gathered: row m -> A[idx[m]])
// W: bf16 [N,K] row-major. C: f32 [M,N].
// Block = 256 threads = 8 waves. Wave w computes a 16x64 strip:
//   rows  = (blockIdx.y*8 + w)*16 .. +15, cols = blockIdx.x*64 .. +63 (4 WMMA tiles)
__global__ __launch_bounds__(256) void gemm_bf16_kernel(
    const __bf16* __restrict__ A, const __bf16* __restrict__ W,
    const float* __restrict__ b1, const float* __restrict__ b2,
    float* __restrict__ C, const int* __restrict__ gidx, int N, int K) {
  const int tid  = threadIdx.x;
  const int wave = tid >> 5;
  const int lane = tid & 31;
  const int half = lane >> 4;
  const int nl   = lane & 15;

  const int mtile = blockIdx.y * 8 + wave;
  const int nbase = blockIdx.x * 64;
  const int mrow  = mtile * 16 + nl;

  const __bf16* arow = gidx ? (A + (size_t)gidx[mrow] * K)
                            : (A + (size_t)mrow * K);
  const __bf16* brow0 = W + (size_t)(nbase + 0  + nl) * K;
  const __bf16* brow1 = W + (size_t)(nbase + 16 + nl) * K;
  const __bf16* brow2 = W + (size_t)(nbase + 32 + nl) * K;
  const __bf16* brow3 = W + (size_t)(nbase + 48 + nl) * K;

  f32x8 acc0 = {}, acc1 = {}, acc2 = {}, acc3 = {};

  for (int k = 0; k < K; k += 32) {
    __builtin_prefetch((const void*)(arow + k + 256), 0, 0);  // global_prefetch_b8
    bf16x16 a  = load_a_frag(arow, k, half);
    bf16x16 v0 = load_b_frag(brow0, k, half);
    bf16x16 v1 = load_b_frag(brow1, k, half);
    bf16x16 v2 = load_b_frag(brow2, k, half);
    bf16x16 v3 = load_b_frag(brow3, k, half);
    acc0 = wmma_bf16(a, v0, acc0);
    acc1 = wmma_bf16(a, v1, acc1);
    acc2 = wmma_bf16(a, v2, acc2);
    acc3 = wmma_bf16(a, v3, acc3);
  }

  f32x8 accs[4] = {acc0, acc1, acc2, acc3};
#pragma unroll
  for (int f = 0; f < 4; ++f) {
    const int n = nbase + f * 16 + nl;
    float bs = b1 ? b1[n] : 0.0f;
    if (b2) bs += b2[n];
#pragma unroll
    for (int r = 0; r < 8; ++r) {
      const int row = mtile * 16 + r + half * 8;
      C[(size_t)row * N + n] = accs[f][r] + bs;
    }
  }
}

// ---------------- grid-wide barrier (monotonic counter) ----------------
static __device__ __forceinline__ void grid_barrier(unsigned* cnt, unsigned target) {
  __syncthreads();
  if (threadIdx.x == 0) {
    __threadfence();
    atomicAdd(cnt, 1u);
    while (__hip_atomic_load(cnt, __ATOMIC_RELAXED, __HIP_MEMORY_SCOPE_AGENT) < target)
      __builtin_amdgcn_s_sleep(2);
  }
  __syncthreads();
  __threadfence();
}

// ---------------- persistent LSTM recurrence for one layer ----------------
// Grid: 64 WGs x 256 threads. WG (rt=blk/16, jt=blk%16) owns batch rows
// [rt*16, rt*16+16) and hidden columns [jt*64, jt*64+64) (=> 256 gate cols).
// Per step: gates = xg[t] + h_prev @ Whh^T via WMMA; then gate nonlinearity,
// c/h update; h written bf16 to the alternate global h buffer; one grid
// barrier per step (double-buffered h removes the read/write hazard).
__global__ __launch_bounds__(256) void lstm_seq_kernel(
    const float* __restrict__ xg,        // [B*S, 4H] f32
    const __bf16* __restrict__ Whh,      // [4H, H] bf16 row-major
    __bf16* __restrict__ hbuf,           // [2, B, H] bf16 (double buffer)
    __bf16* __restrict__ o_out,          // [B, S, H] bf16
    float* __restrict__ hN_out,          // [B, H] f32
    float* __restrict__ cN_out,          // [B, H] f32
    unsigned* __restrict__ bar) {
  __shared__ float g_lds[16 * 256];      // gates: [m(16)][gate(4)*64 + jl]
  __shared__ float c_lds[16 * 64];       // cell state for this WG's tile

  const int tid  = threadIdx.x;
  const int wave = tid >> 5;
  const int lane = tid & 31;
  const int half = lane >> 4;
  const int nl   = lane & 15;
  const int rt   = blockIdx.x >> 4;      // batch-row tile (0..3)
  const int jt   = blockIdx.x & 15;      // hidden-column tile (0..15)
  const size_t HBUF = (size_t)Bb * Hh;

  // init c = 0 and our slice of h buffer 0 = 0
  for (int id = tid; id < 16 * 64; id += 256) {
    c_lds[id] = 0.0f;
    const int m = id >> 6, jl = id & 63;
    hbuf[(size_t)(rt * 16 + m) * Hh + jt * 64 + jl] = (__bf16)0.0f;
  }
  unsigned target = 64u;
  grid_barrier(bar, target);

  // this wave's two 16-col gate tiles
  const int ct0 = wave * 2, ct1 = wave * 2 + 1;
  const int col0 = (ct0 >> 2) * Hh + jt * 64 + (ct0 & 3) * 16;
  const int col1 = (ct1 >> 2) * Hh + jt * 64 + (ct1 & 3) * 16;
  const __bf16* brow0 = Whh + (size_t)(col0 + nl) * Hh;
  const __bf16* brow1 = Whh + (size_t)(col1 + nl) * Hh;

  for (int t = 0; t < Ss; ++t) {
    const __bf16* hprev = hbuf + (size_t)(t & 1) * HBUF;
    const __bf16* arow  = hprev + (size_t)(rt * 16 + nl) * Hh;

    f32x8 acc0 = {}, acc1 = {};
    for (int k = 0; k < Hh; k += 32) {
      bf16x16 a  = load_a_frag(arow, k, half);
      bf16x16 v0 = load_b_frag(brow0, k, half);
      bf16x16 v1 = load_b_frag(brow1, k, half);
      acc0 = wmma_bf16(a, v0, acc0);
      acc1 = wmma_bf16(a, v1, acc1);
    }

    // gates -> LDS (add precomputed input contribution from xg)
    f32x8 accs[2] = {acc0, acc1};
    const int cts[2] = {ct0, ct1};
    const int cols[2] = {col0, col1};
#pragma unroll
    for (int f = 0; f < 2; ++f) {
      const int gate = cts[f] >> 2, jj = cts[f] & 3;
#pragma unroll
      for (int r = 0; r < 8; ++r) {
        const int m = r + half * 8;
        const int brow_g = rt * 16 + m;
        const float v = accs[f][r] +
            xg[((size_t)brow_g * Ss + t) * G4H + cols[f] + nl];
        g_lds[m * 256 + gate * 64 + jj * 16 + nl] = v;
      }
    }
    __syncthreads();

    // elementwise gate math: 16 rows x 64 cols = 1024 items, 4 per thread
#pragma unroll
    for (int q = 0; q < 4; ++q) {
      const int id = tid + q * 256;
      const int m = id >> 6, jl = id & 63;
      const float gi = sigmoidf_(g_lds[m * 256 + 0 * 64 + jl]);
      const float gf = sigmoidf_(g_lds[m * 256 + 1 * 64 + jl]);
      const float gg = tanhf    (g_lds[m * 256 + 2 * 64 + jl]);
      const float go = sigmoidf_(g_lds[m * 256 + 3 * 64 + jl]);
      const float cn = gf * c_lds[id] + gi * gg;
      const float hn = go * tanhf(cn);
      c_lds[id] = cn;
      const int bg = rt * 16 + m, hcol = jt * 64 + jl;
      hbuf[(size_t)((t + 1) & 1) * HBUF + (size_t)bg * Hh + hcol] = (__bf16)hn;
      o_out[((size_t)bg * Ss + t) * Hh + hcol] = (__bf16)hn;
      if (t == Ss - 1) {
        hN_out[(size_t)bg * Hh + hcol] = hn;
        cN_out[(size_t)bg * Hh + hcol] = cn;
      }
    }
    target += 64u;
    grid_barrier(bar, target);
  }
}

// ---------------- host-side orchestration ----------------
extern "C" void kernel_launch(void* const* d_in, const int* in_sizes, int n_in,
                              void* d_out, int out_size, void* d_ws, size_t ws_size,
                              hipStream_t stream) {
  (void)in_sizes; (void)n_in; (void)out_size; (void)ws_size;
  const int*   x    = (const int*)  d_in[0];   // [B,S]
  const float* emb  = (const float*)d_in[1];   // [V,E]
  const float* Wih0 = (const float*)d_in[2];   // [4H,E]
  const float* Whh0 = (const float*)d_in[3];   // [4H,H]
  const float* bih0 = (const float*)d_in[4];
  const float* bhh0 = (const float*)d_in[5];
  const float* Wih1 = (const float*)d_in[6];   // [4H,H]
  const float* Whh1 = (const float*)d_in[7];   // [4H,H]
  const float* bih1 = (const float*)d_in[8];
  const float* bhh1 = (const float*)d_in[9];
  const float* fcW  = (const float*)d_in[10];  // [V,H]
  const float* fcb  = (const float*)d_in[11];

  float* logits = (float*)d_out;                         // [B,S,V]
  float* hN     = logits + (size_t)BS * Vv;              // [2,B,H]
  float* cN     = hN + 2 * (size_t)Bb * Hh;              // [2,B,H]

  // workspace carve-up (256B aligned)
  char* ws = (char*)d_ws;
  size_t off = 0;
  auto alloc = [&](size_t bytes) -> char* {
    char* p = ws + off;
    off = (off + bytes + 255) & ~(size_t)255;
    return p;
  };
  unsigned* bars    = (unsigned*)alloc(256);                          // 2 barrier counters
  __bf16* emb_bf    = (__bf16*)alloc((size_t)Vv * Ee * 2);
  __bf16* Wih0_bf   = (__bf16*)alloc((size_t)G4H * Ee * 2);
  __bf16* Whh0_bf   = (__bf16*)alloc((size_t)G4H * Hh * 2);
  __bf16* Wih1_bf   = (__bf16*)alloc((size_t)G4H * Hh * 2);
  __bf16* Whh1_bf   = (__bf16*)alloc((size_t)G4H * Hh * 2);
  __bf16* fcW_bf    = (__bf16*)alloc((size_t)Vv * Hh * 2);
  __bf16* hbuf      = (__bf16*)alloc((size_t)2 * Bb * Hh * 2);
  __bf16* o0_bf     = (__bf16*)alloc((size_t)BS * Hh * 2);
  __bf16* o1_bf     = (__bf16*)alloc((size_t)BS * Hh * 2);
  float*  xg        = (float*)alloc((size_t)BS * G4H * 4);            // reused for both layers

  zero_counters_kernel<<<1, 64, 0, stream>>>(bars);

  auto cvt = [&](const float* s, __bf16* d, int n) {
    int blocks = (n + 255) / 256; if (blocks > 2048) blocks = 2048;
    f32_to_bf16_kernel<<<blocks, 256, 0, stream>>>(s, d, n);
  };
  cvt(emb,  emb_bf,  Vv * Ee);
  cvt(Wih0, Wih0_bf, G4H * Ee);
  cvt(Whh0, Whh0_bf, G4H * Hh);
  cvt(Wih1, Wih1_bf, G4H * Hh);
  cvt(Whh1, Whh1_bf, G4H * Hh);
  cvt(fcW,  fcW_bf,  Vv * Hh);

  // layer 0: xg0 = gather(emb, x) @ Wih0^T + bih0 + bhh0
  gemm_bf16_kernel<<<dim3(G4H / 64, BS / 128), 256, 0, stream>>>(
      emb_bf, Wih0_bf, bih0, bhh0, xg, x, G4H, Ee);
  lstm_seq_kernel<<<64, 256, 0, stream>>>(
      xg, Whh0_bf, hbuf, o0_bf, hN, cN, bars + 0);

  // layer 1: xg1 = o0 @ Wih1^T + bih1 + bhh1
  gemm_bf16_kernel<<<dim3(G4H / 64, BS / 128), 256, 0, stream>>>(
      o0_bf, Wih1_bf, bih1, bhh1, xg, nullptr, G4H, Hh);
  lstm_seq_kernel<<<64, 256, 0, stream>>>(
      xg, Whh1_bf, hbuf, o1_bf, hN + (size_t)Bb * Hh, cN + (size_t)Bb * Hh, bars + 1);

  // logits = o1 @ fcW^T + fcb
  gemm_bf16_kernel<<<dim3(Vv / 64, BS / 128), 256, 0, stream>>>(
      o1_bf, fcW_bf, fcb, nullptr, logits, nullptr, Vv, Hh);
}
